// TacotronGST_11699490914874
// MI455X (gfx1250) — matrix operand order repaired and checked by reading, not compiled, not run on hardware
//
#include <hip/hip_runtime.h>
#include <hip/hip_bf16.h>

// Problem constants (match reference)
#define B_    64
#define S_    512
#define T_    500
#define EMB_  512
#define ENC_  512
#define H_    256      // per-direction encoder LSTM hidden
#define DEC_  1024
#define MEL_  80
#define MELP_ 96       // MEL padded to a multiple of 32 for WMMA K-loop
#define QM_   592      // concat attn-query (512) + mel (80) output cols
#define NTOK_ 10
#define NHEAD_ 8

typedef unsigned short bf16_t;  // raw bf16 bits
typedef __attribute__((ext_vector_type(16))) __bf16 v16bf;
typedef __attribute__((ext_vector_type(8)))  float  v8f;

union FragU { unsigned int u32[8]; v16bf v; };

__device__ __forceinline__ bf16_t f2bf(float f) {
  unsigned u = __float_as_uint(f);
  u += 0x7fffu + ((u >> 16) & 1u);          // round-to-nearest-even
  return (bf16_t)(u >> 16);
}
__device__ __forceinline__ float sigf(float x) { return 1.f / (1.f + __expf(-x)); }

// ---------------------------------------------------------------------------
// Generic WMMA GEMM: Cout[M,N] = A[M,K](bf16) @ W[N,K](bf16)^T (+bias[N])
// One wave owns NB consecutive 16x16 N-tiles (A-fragment reused NB times).
// Fragment layouts per CDNA5 ISA 7.12.2 (wave32).
// ---------------------------------------------------------------------------
template<int NB>
__global__ __launch_bounds__(256)
void wmma_gemm_kernel(const bf16_t* __restrict__ A, int lda,
                      const bf16_t* __restrict__ W, int ldw,
                      const float* __restrict__ bias,
                      float* __restrict__ Cout, long long ldco,
                      int M, int N, int K)
{
  const int wave = blockIdx.x * 8 + (threadIdx.x >> 5);
  const int lane = threadIdx.x & 31;
  const int ntg = (N >> 4) / NB;
  const int mt = wave / ntg;
  const int ng = wave - mt * ntg;
  if (mt * 16 >= M) return;
  const int half = lane >> 4;
  const int l16  = lane & 15;
  const int n0   = ng * NB * 16;
  const bf16_t* arow = A + (long long)(mt * 16 + l16) * lda;
  long long woff[NB];
#pragma unroll
  for (int i = 0; i < NB; ++i) woff[i] = (long long)(n0 + i * 16 + l16) * ldw;
  v8f zero = {};
  v8f acc[NB];
#pragma unroll
  for (int i = 0; i < NB; ++i) acc[i] = zero;
  for (int k0 = 0; k0 < K; k0 += 32) {
    FragU a;
#pragma unroll
    for (int v = 0; v < 8; ++v) {
      // A 16x32: lane-half 0 holds K 0..7,16..23 ; half 1 holds K 8..15,24..31
      int ka = k0 + (v < 4 ? 2 * v : 2 * v + 8) + half * 8;
      a.u32[v] = *(const unsigned int*)(arow + ka);
    }
#pragma unroll
    for (int i = 0; i < NB; ++i) {
      FragU b;
#pragma unroll
      for (int v = 0; v < 8; ++v)   // B 32x16: same K order as A (half 0: K 0..7,16..23; half 1: K 8..15,24..31)
        b.u32[v] = *(const unsigned int*)(W + woff[i] + k0 + (v < 4 ? 2 * v : 2 * v + 8) + half * 8);
      acc[i] = __builtin_amdgcn_wmma_f32_16x16x32_bf16(false, a.v, false, b.v,
                                                       (short)0, acc[i], false, false);
    }
  }
#pragma unroll
  for (int i = 0; i < NB; ++i) {
#pragma unroll
    for (int r = 0; r < 8; ++r) {
      int row = mt * 16 + r + half * 8;
      int col = n0 + i * 16 + l16;
      float val = acc[i][r];
      if (bias) val += bias[col];
      Cout[(long long)row * ldco + col] = val;
    }
  }
}

template<int NB>
static inline void gemmN(hipStream_t st, const bf16_t* A, int lda,
                         const bf16_t* W, int ldw, const float* bias,
                         float* Cout, long long ldco, int M, int N, int K) {
  int waves = (M / 16) * ((N / 16) / NB);
  wmma_gemm_kernel<NB><<<(waves + 7) / 8, 256, 0, st>>>(A, lda, W, ldw, bias,
                                                        Cout, ldco, M, N, K);
}

// ---------------------------------------------------------------------------
// Fused LSTM step: WMMA gates (+optional second input GEMM) -> LDS -> update.
// Block (mt,jg) owns batch rows [16mt,16mt+16) and hidden cols [jg*JGW,+JGW),
// computing all 4 gate quarters for that slice so the nonlinearity is local.
// h is double-buffered across steps (hprev read-only here, hout written).
// ---------------------------------------------------------------------------
template<int HID, int JGW, int KIN>
__global__ __launch_bounds__(256)
void k_fused_lstm(const bf16_t* __restrict__ hprev,   // [B,HID] bf16
                  const bf16_t* __restrict__ Wrec,    // [4*HID,HID] bf16
                  const bf16_t* __restrict__ A2,      // [B,KIN] bf16 (or null)
                  const bf16_t* __restrict__ W2,      // [4*HID,KIN] bf16
                  const float* __restrict__ Cin, long long ldci, // per-row gate base
                  float* __restrict__ c,              // [B,HID] state
                  bf16_t* __restrict__ hout,          // [B,HID] bf16
                  float* __restrict__ hout_f,         // [B,HID] f32 (or null)
                  float* __restrict__ encO, int s, int coloff)   // enc sink (or null)
{
  constexpr int NT16 = JGW / 16;     // N-tiles per gate quarter in this block
  constexpr int TPW  = (4 * NT16) / 8; // tiles per wave (8 waves)
  const int JGB = HID / JGW;
  const int mt = blockIdx.x / JGB;
  const int jg = blockIdx.x % JGB;
  const int wave = threadIdx.x >> 5;
  const int lane = threadIdx.x & 31;
  const int half = lane >> 4, l16 = lane & 15;

  __shared__ float lds[4][16][JGW];

  const bf16_t* arow = hprev + (long long)(mt * 16 + l16) * HID;
  int ncol[TPW];
#pragma unroll
  for (int i = 0; i < TPW; ++i) {
    int t = wave * TPW + i;
    int r = t / NT16, nt = t % NT16;
    ncol[i] = r * HID + jg * JGW + nt * 16;   // global gate column base
  }
  v8f zero = {};
  v8f acc[TPW];
#pragma unroll
  for (int i = 0; i < TPW; ++i) acc[i] = zero;

  for (int k0 = 0; k0 < HID; k0 += 32) {
    FragU a;
#pragma unroll
    for (int v = 0; v < 8; ++v) {
      int ka = k0 + (v < 4 ? 2 * v : 2 * v + 8) + half * 8;
      a.u32[v] = *(const unsigned int*)(arow + ka);
    }
#pragma unroll
    for (int i = 0; i < TPW; ++i) {
      FragU b;
#pragma unroll
      for (int v = 0; v < 8; ++v)
        b.u32[v] = *(const unsigned int*)(Wrec + (long long)(ncol[i] + l16) * HID
                                          + k0 + (v < 4 ? 2 * v : 2 * v + 8) + half * 8);
      acc[i] = __builtin_amdgcn_wmma_f32_16x16x32_bf16(false, a.v, false, b.v,
                                                       (short)0, acc[i], false, false);
    }
  }
  if constexpr (KIN > 0) {
    const bf16_t* a2row = A2 + (long long)(mt * 16 + l16) * KIN;
    for (int k0 = 0; k0 < KIN; k0 += 32) {
      FragU a;
#pragma unroll
      for (int v = 0; v < 8; ++v) {
        int ka = k0 + (v < 4 ? 2 * v : 2 * v + 8) + half * 8;
        a.u32[v] = *(const unsigned int*)(a2row + ka);
      }
#pragma unroll
      for (int i = 0; i < TPW; ++i) {
        FragU b;
#pragma unroll
        for (int v = 0; v < 8; ++v)
          b.u32[v] = *(const unsigned int*)(W2 + (long long)(ncol[i] + l16) * KIN
                                            + k0 + (v < 4 ? 2 * v : 2 * v + 8) + half * 8);
        acc[i] = __builtin_amdgcn_wmma_f32_16x16x32_bf16(false, a.v, false, b.v,
                                                         (short)0, acc[i], false, false);
      }
    }
  }
  // spill gate tiles (+ per-row gate base) to LDS
#pragma unroll
  for (int i = 0; i < TPW; ++i) {
    int t = wave * TPW + i;
    int r = t / NT16, nt = t % NT16;
#pragma unroll
    for (int rr = 0; rr < 8; ++rr) {
      int lm  = rr + half * 8;
      int row = mt * 16 + lm;
      lds[r][lm][nt * 16 + l16] =
          acc[i][rr] + Cin[(long long)row * ldci + ncol[i] + l16];
    }
  }
  __syncthreads();
  // gate nonlinearity + state update (slice owned exclusively by this block)
  for (int u = threadIdx.x; u < 16 * JGW; u += 256) {
    int lm = u / JGW, jj = u % JGW;
    int b = mt * 16 + lm;
    int j = jg * JGW + jj;
    float gi = sigf(lds[0][lm][jj]);
    float gf = sigf(lds[1][lm][jj]);
    float gg = tanhf(lds[2][lm][jj]);
    float go = sigf(lds[3][lm][jj]);
    long long idx = (long long)b * HID + j;
    float cn = gf * c[idx] + gi * gg;
    float hv = go * tanhf(cn);
    c[idx] = cn;
    hout[idx] = f2bf(hv);
    if (hout_f) hout_f[idx] = hv;
    if (encO) encO[((long long)b * S_ + s) * ENC_ + coloff + j] = hv;
  }
}

// ---------------------------------------------------------------------------
// Small helper kernels
// ---------------------------------------------------------------------------
__global__ void k_zero_f(float* p, long long n) {
  long long i = (long long)blockIdx.x * blockDim.x + threadIdx.x;
  if (i < n) p[i] = 0.f;
}
__global__ void k_zero_u16(bf16_t* p, long long n) {
  long long i = (long long)blockIdx.x * blockDim.x + threadIdx.x;
  if (i < n) p[i] = 0;
}
__global__ void k_biasadd(const float* a, const float* b, float* o, int n) {
  int i = blockIdx.x * blockDim.x + threadIdx.x;
  if (i < n) o[i] = a[i] + b[i];
}
__global__ void k_concat_bias(const float* a, const float* b, float* o) {
  int i = blockIdx.x * blockDim.x + threadIdx.x;
  if (i < QM_) o[i] = (i < ENC_) ? a[i] : b[i - ENC_];
}
// submatrix f32 -> bf16 with optional zero-padded columns
__global__ void k_sub2bf(const float* __restrict__ src, int src_ld, int src_col0,
                         bf16_t* __restrict__ dst, int dst_ld, int rows, int copy_cols) {
  long long idx = (long long)blockIdx.x * blockDim.x + threadIdx.x;
  long long tot = (long long)rows * dst_ld;
  if (idx >= tot) return;
  int r = (int)(idx / dst_ld), c = (int)(idx % dst_ld);
  float v = (c < copy_cols) ? src[(long long)r * src_ld + src_col0 + c] : 0.f;
  dst[idx] = f2bf(v);
}
__global__ void k_gather_emb(const int* __restrict__ text, const float* __restrict__ tbl,
                             bf16_t* __restrict__ dst) {
  long long idx = (long long)blockIdx.x * blockDim.x + threadIdx.x;
  if (idx >= (long long)B_ * S_ * EMB_) return;
  int e  = (int)(idx & (EMB_ - 1));
  int bs = (int)(idx >> 9);
  dst[idx] = f2bf(tbl[(long long)text[bs] * EMB_ + e]);
}

// ---------------------------------------------------------------------------
// GST: fold MHA weights so scores = enc . kq + kb   (one GEMM, no q tensor)
// ---------------------------------------------------------------------------
__global__ void k_gst_k(const float* __restrict__ tokens, const float* __restrict__ k_w,
                        const float* __restrict__ k_b, float* __restrict__ ktok) {
  int idx = blockIdx.x * blockDim.x + threadIdx.x;     // 10*512
  if (idx >= NTOK_ * ENC_) return;
  int t = idx / ENC_, nh = idx % ENC_;
  float s = k_b[nh];
  const float* tr = tokens + (long long)t * ENC_;
  const float* wr = k_w + (long long)nh * ENC_;
  for (int e = 0; e < ENC_; ++e) s += tr[e] * wr[e];
  ktok[idx] = s;
}
__global__ void k_gst_kb(const float* __restrict__ ktok, const float* __restrict__ q_b,
                         float* __restrict__ kb) {
  int p = blockIdx.x * blockDim.x + threadIdx.x;       // 80
  if (p >= NHEAD_ * NTOK_) return;
  int n = p / NTOK_, t = p % NTOK_;
  float s = 0.f;
  for (int h = 0; h < 64; ++h) s += ktok[t * ENC_ + n * 64 + h] * q_b[n * 64 + h];
  kb[p] = 0.125f * s;
}
__global__ void k_gst_kq(const float* __restrict__ ktok, const float* __restrict__ q_w,
                         bf16_t* __restrict__ kq) {
  int idx = blockIdx.x * blockDim.x + threadIdx.x;     // 80*512
  if (idx >= NHEAD_ * NTOK_ * ENC_) return;
  int p = idx / ENC_, e = idx % ENC_;
  int n = p / NTOK_, t = p % NTOK_;
  float s = 0.f;
  for (int h = 0; h < 64; ++h)
    s += ktok[t * ENC_ + n * 64 + h] * q_w[(long long)(n * 64 + h) * ENC_ + e];
  kq[idx] = f2bf(0.125f * s);
}
__global__ void k_gst_softmax(const float* __restrict__ scores, float* __restrict__ sw) {
  int b = blockIdx.x, s = threadIdx.x;                 // 64 blocks x 512 threads
  __shared__ float acc[NTOK_];
  if (s < NTOK_) acc[s] = 0.f;
  __syncthreads();
  const float* sc = scores + ((long long)b * S_ + s) * (NHEAD_ * NTOK_);
  float w[NTOK_];
#pragma unroll
  for (int t = 0; t < NTOK_; ++t) w[t] = 0.f;
  for (int n = 0; n < NHEAD_; ++n) {
    const float* p = sc + n * NTOK_;
    float mx = p[0];
    for (int t = 1; t < NTOK_; ++t) mx = fmaxf(mx, p[t]);
    float e[NTOK_], ssum = 0.f;
    for (int t = 0; t < NTOK_; ++t) { e[t] = __expf(p[t] - mx); ssum += e[t]; }
    float inv = 1.f / ssum;
    for (int t = 0; t < NTOK_; ++t) w[t] += e[t] * inv;
  }
  for (int t = 0; t < NTOK_; ++t) atomicAdd(&acc[t], w[t] * (1.f / NHEAD_));
  __syncthreads();
  if (s < NTOK_) sw[b * NTOK_ + s] = acc[s] * (1.f / S_);
}
__global__ void k_style1(const float* __restrict__ sw, const float* __restrict__ tokens,
                         float* __restrict__ sv) {
  int idx = blockIdx.x * blockDim.x + threadIdx.x;     // 64*512
  if (idx >= B_ * ENC_) return;
  int b = idx >> 9, e = idx & (ENC_ - 1);
  float s = 0.f;
  for (int t = 0; t < NTOK_; ++t) s += sw[b * NTOK_ + t] * tokens[t * ENC_ + e];
  sv[idx] = s;
}
__global__ void k_style2(const float* __restrict__ sv, const float* __restrict__ proj_w,
                         const float* __restrict__ proj_b, float* __restrict__ style) {
  int idx = blockIdx.x * blockDim.x + threadIdx.x;     // 64*512
  if (idx >= B_ * ENC_) return;
  int b = idx >> 9, e = idx & (ENC_ - 1);
  float s = proj_b[e];
  const float* sr = sv + (long long)b * ENC_;
  const float* pr = proj_w + (long long)e * ENC_;
  for (int i = 0; i < ENC_; ++i) s += sr[i] * pr[i];
  style[idx] = s;
}
__global__ void k_addstyle(float* __restrict__ enc, const float* __restrict__ style) {
  long long idx = (long long)blockIdx.x * blockDim.x + threadIdx.x;
  if (idx >= (long long)B_ * S_ * ENC_) return;
  int e = (int)(idx & (ENC_ - 1));
  int b = (int)(idx / ((long long)S_ * ENC_));
  enc[idx] += style[b * ENC_ + e];
}
__global__ void k_encmean(const float* __restrict__ enc, float* __restrict__ mean,
                          bf16_t* __restrict__ mean_bf) {
  int b = blockIdx.x, e = threadIdx.x;                 // 64 x 512
  float s = 0.f;
  for (int t = 0; t < S_; ++t) s += enc[((long long)b * S_ + t) * ENC_ + e];
  float m = s * (1.f / S_);
  mean[b * ENC_ + e] = m;
  mean_bf[b * ENC_ + e] = f2bf(m);
}
// decoder teacher-forcing inputs (t=0 -> zeros; padded to 96 cols) in bf16
__global__ void k_melprev(const float* __restrict__ mel_target, bf16_t* __restrict__ dst) {
  long long idx = (long long)blockIdx.x * blockDim.x + threadIdx.x;
  if (idx >= (long long)T_ * B_ * MELP_) return;
  int t = (int)(idx / (B_ * MELP_));
  int r = (int)(idx % (B_ * MELP_));
  int b = r / MELP_, m = r % MELP_;
  float v = (t == 0 || m >= MEL_) ? 0.f
          : mel_target[((long long)b * T_ + (t - 1)) * MEL_ + m];
  dst[idx] = f2bf(v);
}

// fused decoder attention softmax (enc stays in L2 as bf16) + mel copy + stop
__global__ void k_attn_stop(const float* __restrict__ qm, const bf16_t* __restrict__ encb,
                            const float* __restrict__ h, const float* __restrict__ stop_w,
                            const float* __restrict__ stop_b,
                            float* __restrict__ out_mel, float* __restrict__ out_attn,
                            float* __restrict__ out_stop, int t) {
  int b = blockIdx.x, tid = threadIdx.x;               // 64 blocks x 512 threads
  __shared__ float qs[ENC_];
  __shared__ float red[512];
  qs[tid] = qm[b * QM_ + tid];
  if (tid < MEL_)
    out_mel[((long long)b * T_ + t) * MEL_ + tid] = qm[b * QM_ + ENC_ + tid];
  __syncthreads();
  const bf16_t* er = encb + ((long long)b * S_ + tid) * ENC_;
  float sc = 0.f;
  for (int e = 0; e < ENC_; e += 2) {
    unsigned u = *(const unsigned*)(er + e);
    sc += qs[e]     * __uint_as_float(u << 16);
    sc += qs[e + 1] * __uint_as_float(u & 0xffff0000u);
  }
  red[tid] = sc; __syncthreads();
  for (int off = 256; off; off >>= 1) {
    if (tid < off) red[tid] = fmaxf(red[tid], red[tid + off]);
    __syncthreads();
  }
  float mx = red[0]; __syncthreads();
  float ex = __expf(sc - mx);
  red[tid] = ex; __syncthreads();
  for (int off = 256; off; off >>= 1) {
    if (tid < off) red[tid] += red[tid + off];
    __syncthreads();
  }
  float inv = 1.f / red[0];
  out_attn[((long long)b * T_ + t) * S_ + tid] = ex * inv;
  __syncthreads();
  float part = 0.f;
  if (tid < 256)
    for (int k = tid; k < DEC_; k += 256) part += h[b * DEC_ + k] * stop_w[k];
  red[tid] = part; __syncthreads();
  for (int off = 256; off; off >>= 1) {
    if (tid < off) red[tid] += red[tid + off];
    __syncthreads();
  }
  if (tid == 0) out_stop[b * T_ + t] = 1.f / (1.f + __expf(-(red[0] + stop_b[0])));
}

// ---------------------------------------------------------------------------
static inline int gb(long long n, int t) { return (int)((n + t - 1) / t); }

extern "C" void kernel_launch(void* const* d_in, const int* in_sizes, int n_in,
                              void* d_out, int out_size, void* d_ws, size_t ws_size,
                              hipStream_t stream)
{
  (void)in_sizes; (void)n_in; (void)out_size; (void)ws_size;
  const int*   text       = (const int*)  d_in[0];
  const float* mel_target = (const float*)d_in[1];
  const float* emb_table  = (const float*)d_in[3];
  const float* Wih_f = (const float*)d_in[4],  *Whh_f = (const float*)d_in[5];
  const float* bih_f = (const float*)d_in[6],  *bhh_f = (const float*)d_in[7];
  const float* Wih_b = (const float*)d_in[8],  *Whh_b = (const float*)d_in[9];
  const float* bih_b = (const float*)d_in[10], *bhh_b = (const float*)d_in[11];
  const float* tokens = (const float*)d_in[12];
  const float* q_w = (const float*)d_in[13], *q_b = (const float*)d_in[14];
  const float* k_w = (const float*)d_in[15], *k_b = (const float*)d_in[16];
  const float* proj_w = (const float*)d_in[17], *proj_b = (const float*)d_in[18];
  const float* dWih = (const float*)d_in[19], *dWhh = (const float*)d_in[20];
  const float* dbih = (const float*)d_in[21], *dbhh = (const float*)d_in[22];
  const float* attn_w = (const float*)d_in[23], *attn_b = (const float*)d_in[24];
  const float* mel_w = (const float*)d_in[25], *mel_b = (const float*)d_in[26];
  const float* stop_w = (const float*)d_in[27], *stop_b = (const float*)d_in[28];

  float* out_mel  = (float*)d_out;
  float* out_attn = out_mel + (size_t)B_ * T_ * MEL_;
  float* out_stop = out_attn + (size_t)B_ * T_ * S_;

  // bump allocator over workspace
  char* ws = (char*)d_ws; size_t off = 0;
  auto alloc = [&](size_t bytes) -> char* {
    size_t a = (off + 255) & ~(size_t)255; off = a + bytes; return ws + a;
  };
  bf16_t* emb_bf    = (bf16_t*)alloc((size_t)B_ * S_ * EMB_ * 2);
  float*  pre       = (float*) alloc((size_t)B_ * S_ * 4 * H_ * 4);  // reused fwd/bwd
  float*  enc       = (float*) alloc((size_t)B_ * S_ * ENC_ * 4);
  bf16_t* enc_bf    = (bf16_t*)alloc((size_t)B_ * S_ * ENC_ * 2);
  bf16_t* wihf_bf   = (bf16_t*)alloc((size_t)4 * H_ * EMB_ * 2);
  bf16_t* wihb_bf   = (bf16_t*)alloc((size_t)4 * H_ * EMB_ * 2);
  bf16_t* whhf_bf   = (bf16_t*)alloc((size_t)4 * H_ * H_ * 2);
  bf16_t* whhb_bf   = (bf16_t*)alloc((size_t)4 * H_ * H_ * 2);
  float*  ebias_f   = (float*) alloc(4 * H_ * 4);
  float*  ebias_b   = (float*) alloc(4 * H_ * 4);
  bf16_t* h_enc_bf0 = (bf16_t*)alloc((size_t)B_ * H_ * 2);
  bf16_t* h_enc_bf1 = (bf16_t*)alloc((size_t)B_ * H_ * 2);
  float*  c_enc     = (float*) alloc((size_t)B_ * H_ * 4);
  float*  ktok      = (float*) alloc((size_t)NTOK_ * ENC_ * 4);
  float*  kb        = (float*) alloc(NHEAD_ * NTOK_ * 4);
  bf16_t* kq_bf     = (bf16_t*)alloc((size_t)NHEAD_ * NTOK_ * ENC_ * 2);
  float*  scores    = (float*) alloc((size_t)B_ * S_ * NHEAD_ * NTOK_ * 4);
  float*  sw        = (float*) alloc(B_ * NTOK_ * 4);
  float*  sv        = (float*) alloc((size_t)B_ * ENC_ * 4);
  float*  stylev    = (float*) alloc((size_t)B_ * ENC_ * 4);
  float*  enc_mean  = (float*) alloc((size_t)B_ * ENC_ * 4);
  bf16_t* emean_bf  = (bf16_t*)alloc((size_t)B_ * ENC_ * 2);
  float*  dbias     = (float*) alloc(4 * DEC_ * 4);
  bf16_t* dwmel_bf  = (bf16_t*)alloc((size_t)4 * DEC_ * MELP_ * 2);
  bf16_t* dwenc_bf  = (bf16_t*)alloc((size_t)4 * DEC_ * ENC_ * 2);
  bf16_t* dwhh_bf   = (bf16_t*)alloc((size_t)4 * DEC_ * DEC_ * 2);
  bf16_t* qmw_bf    = (bf16_t*)alloc((size_t)QM_ * DEC_ * 2);
  float*  qmb       = (float*) alloc(QM_ * 4);
  float*  cbase     = (float*) alloc((size_t)B_ * 4 * DEC_ * 4);
  bf16_t* melp_bf   = (bf16_t*)alloc((size_t)T_ * B_ * MELP_ * 2);
  float*  h_dec     = (float*) alloc((size_t)B_ * DEC_ * 4);
  float*  c_dec     = (float*) alloc((size_t)B_ * DEC_ * 4);
  bf16_t* h_dec_bf0 = (bf16_t*)alloc((size_t)B_ * DEC_ * 2);
  bf16_t* h_dec_bf1 = (bf16_t*)alloc((size_t)B_ * DEC_ * 2);
  float*  qm        = (float*) alloc((size_t)B_ * QM_ * 4);

  // ---- weight / input conversions --------------------------------------
  k_gather_emb<<<gb((long long)B_ * S_ * EMB_, 256), 256, 0, stream>>>(text, emb_table, emb_bf);
  k_sub2bf<<<gb((long long)4 * H_ * EMB_, 256), 256, 0, stream>>>(Wih_f, EMB_, 0, wihf_bf, EMB_, 4 * H_, EMB_);
  k_sub2bf<<<gb((long long)4 * H_ * EMB_, 256), 256, 0, stream>>>(Wih_b, EMB_, 0, wihb_bf, EMB_, 4 * H_, EMB_);
  k_sub2bf<<<gb((long long)4 * H_ * H_, 256), 256, 0, stream>>>(Whh_f, H_, 0, whhf_bf, H_, 4 * H_, H_);
  k_sub2bf<<<gb((long long)4 * H_ * H_, 256), 256, 0, stream>>>(Whh_b, H_, 0, whhb_bf, H_, 4 * H_, H_);
  k_biasadd<<<gb(4 * H_, 256), 256, 0, stream>>>(bih_f, bhh_f, ebias_f, 4 * H_);
  k_biasadd<<<gb(4 * H_, 256), 256, 0, stream>>>(bih_b, bhh_b, ebias_b, 4 * H_);

  // ---- forward LSTM (pre-projection batched; recurrence fused/step) -----
  gemmN<4>(stream, emb_bf, EMB_, wihf_bf, EMB_, ebias_f, pre, 4 * H_, B_ * S_, 4 * H_, EMB_);
  k_zero_u16<<<gb(B_ * H_, 256), 256, 0, stream>>>(h_enc_bf0, B_ * H_);
  k_zero_f<<<gb(B_ * H_, 256), 256, 0, stream>>>(c_enc, B_ * H_);
  {
    bf16_t* hp = h_enc_bf0; bf16_t* hc = h_enc_bf1;
    for (int s = 0; s < S_; ++s) {
      k_fused_lstm<H_, 128, 0><<<4 * (H_ / 128), 256, 0, stream>>>(
          hp, whhf_bf, nullptr, nullptr, pre + (long long)s * 4 * H_,
          (long long)S_ * 4 * H_, c_enc, hc, nullptr, enc, s, 0);
      bf16_t* tmp = hp; hp = hc; hc = tmp;
    }
  }
  // ---- backward LSTM ----------------------------------------------------
  gemmN<4>(stream, emb_bf, EMB_, wihb_bf, EMB_, ebias_b, pre, 4 * H_, B_ * S_, 4 * H_, EMB_);
  k_zero_u16<<<gb(B_ * H_, 256), 256, 0, stream>>>(h_enc_bf0, B_ * H_);
  k_zero_f<<<gb(B_ * H_, 256), 256, 0, stream>>>(c_enc, B_ * H_);
  {
    bf16_t* hp = h_enc_bf0; bf16_t* hc = h_enc_bf1;
    for (int s = S_ - 1; s >= 0; --s) {
      k_fused_lstm<H_, 128, 0><<<4 * (H_ / 128), 256, 0, stream>>>(
          hp, whhb_bf, nullptr, nullptr, pre + (long long)s * 4 * H_,
          (long long)S_ * 4 * H_, c_enc, hc, nullptr, enc, s, H_);
      bf16_t* tmp = hp; hp = hc; hc = tmp;
    }
  }

  // ---- GST attention (folded) ------------------------------------------
  k_gst_k<<<gb(NTOK_ * ENC_, 256), 256, 0, stream>>>(tokens, k_w, k_b, ktok);
  k_gst_kb<<<1, 128, 0, stream>>>(ktok, q_b, kb);
  k_gst_kq<<<gb(NHEAD_ * NTOK_ * ENC_, 256), 256, 0, stream>>>(ktok, q_w, kq_bf);
  k_sub2bf<<<gb((long long)B_ * S_ * ENC_, 256), 256, 0, stream>>>(enc, ENC_, 0, enc_bf, ENC_, B_ * S_, ENC_);
  gemmN<5>(stream, enc_bf, ENC_, kq_bf, ENC_, kb, scores, NHEAD_ * NTOK_,
           B_ * S_, NHEAD_ * NTOK_, ENC_);
  k_gst_softmax<<<B_, S_, 0, stream>>>(scores, sw);
  k_style1<<<gb(B_ * ENC_, 256), 256, 0, stream>>>(sw, tokens, sv);
  k_style2<<<gb(B_ * ENC_, 256), 256, 0, stream>>>(sv, proj_w, proj_b, stylev);
  k_addstyle<<<gb((long long)B_ * S_ * ENC_, 256), 256, 0, stream>>>(enc, stylev);
  k_encmean<<<B_, ENC_, 0, stream>>>(enc, enc_mean, emean_bf);
  k_sub2bf<<<gb((long long)B_ * S_ * ENC_, 256), 256, 0, stream>>>(enc, ENC_, 0, enc_bf, ENC_, B_ * S_, ENC_);

  // ---- decoder prep -----------------------------------------------------
  k_biasadd<<<gb(4 * DEC_, 256), 256, 0, stream>>>(dbih, dbhh, dbias, 4 * DEC_);
  k_sub2bf<<<gb((long long)4 * DEC_ * MELP_, 256), 256, 0, stream>>>(dWih, ENC_ + MEL_, 0, dwmel_bf, MELP_, 4 * DEC_, MEL_);
  k_sub2bf<<<gb((long long)4 * DEC_ * ENC_, 256), 256, 0, stream>>>(dWih, ENC_ + MEL_, MEL_, dwenc_bf, ENC_, 4 * DEC_, ENC_);
  k_sub2bf<<<gb((long long)4 * DEC_ * DEC_, 256), 256, 0, stream>>>(dWhh, DEC_, 0, dwhh_bf, DEC_, 4 * DEC_, DEC_);
  k_sub2bf<<<gb((long long)ENC_ * DEC_, 256), 256, 0, stream>>>(attn_w, DEC_, 0, qmw_bf, DEC_, ENC_, DEC_);
  k_sub2bf<<<gb((long long)MEL_ * DEC_, 256), 256, 0, stream>>>(mel_w, DEC_, 0, qmw_bf + (size_t)ENC_ * DEC_, DEC_, MEL_, DEC_);
  k_concat_bias<<<gb(QM_, 256), 256, 0, stream>>>(attn_b, mel_b, qmb);
  // constant part of input gates: enc_mean @ dWih_enc^T + dbih + dbhh
  gemmN<4>(stream, emean_bf, ENC_, dwenc_bf, ENC_, dbias, cbase, 4 * DEC_, B_, 4 * DEC_, ENC_);
  k_melprev<<<gb((long long)T_ * B_ * MELP_, 256), 256, 0, stream>>>(mel_target, melp_bf);
  k_zero_u16<<<gb(B_ * DEC_, 256), 256, 0, stream>>>(h_dec_bf0, B_ * DEC_);
  k_zero_f<<<gb(B_ * DEC_, 256), 256, 0, stream>>>(c_dec, B_ * DEC_);
  k_zero_f<<<gb(B_ * DEC_, 256), 256, 0, stream>>>(h_dec, B_ * DEC_);

  // ---- decoder scan: 3 launches/step on the sequential critical path ----
  {
    bf16_t* hp = h_dec_bf0; bf16_t* hc = h_dec_bf1;
    for (int t = 0; t < T_; ++t) {
      k_fused_lstm<DEC_, 128, MELP_><<<4 * (DEC_ / 128), 256, 0, stream>>>(
          hp, dwhh_bf, melp_bf + (long long)t * B_ * MELP_, dwmel_bf,
          cbase, 4 * DEC_, c_dec, hc, h_dec, nullptr, 0, 0);
      gemmN<1>(stream, hc, DEC_, qmw_bf, DEC_, qmb, qm, QM_, B_, QM_, DEC_);
      k_attn_stop<<<B_, S_, 0, stream>>>(qm, enc_bf, h_dec, stop_w, stop_b,
                                         out_mel, out_attn, out_stop, t);
      bf16_t* tmp = hp; hp = hc; hc = tmp;
    }
  }
}
